// HFLongFormerSelfAttentionBlock_17592186044425
// MI455X (gfx1250) — matrix operand order repaired and checked by reading, not compiled, not run on hardware
//
#include <hip/hip_runtime.h>

// ---------------------------------------------------------------------------
// MI455X (gfx1250) Longformer block.
//  - all GEMM work via v_wmma_f32_16x16x32_bf16 (fp32 accumulate)
//  - GEMM K-tiles double-buffered with GLOBAL_LOAD_ASYNC_TO_LDS_B128 (ASYNCcnt)
//  - attention V tile staged by the Tensor Data Mover (TENSORcnt)
// ---------------------------------------------------------------------------

typedef __attribute__((ext_vector_type(16))) __bf16 v16bf;
typedef __attribute__((ext_vector_type(8)))  float  v8f;
typedef __attribute__((ext_vector_type(4)))  unsigned int v4u;
typedef __attribute__((ext_vector_type(8)))  int v8i;
typedef __attribute__((ext_vector_type(4)))  int v4i;

using u16 = unsigned short;

constexpr int B_    = 2;
constexpr int L_    = 4096;
constexpr int D_    = 1024;
constexpr int H_    = 16;
constexpr int WIN_  = 256;
constexpr int EXP_  = 4;
constexpr int F_    = EXP_ * D_;           // 4096
constexpr int DH_   = D_ / H_;             // 64
constexpr int CC_   = WIN_ / 2;            // 128
constexpr int PAD_  = WIN_ - (L_ % WIN_);  // 256
constexpr int S_    = L_ + PAD_;           // 4352
constexpr int NBLK_ = S_ / CC_;            // 34
constexpr int W3_   = 3 * CC_;             // 384

// --------------------------- helpers ---------------------------------------

__device__ __forceinline__ u16 f2bf(float f) {
  union { float f; unsigned int u; } x; x.f = f;
  unsigned int r = x.u + 0x7FFFu + ((x.u >> 16) & 1u);   // RNE
  return (u16)(r >> 16);
}

union Frag {           // one 16x32 bf16 A-fragment or 32x16 bf16 B-fragment
  v16bf v;
  uint4 u[2];
  u16   s[16];
};

__device__ __forceinline__ v8f wmma_bf16(const v16bf& a, const v16bf& b, const v8f& c) {
  return __builtin_amdgcn_wmma_f32_16x16x32_bf16(
      /*neg_a=*/false, a, /*neg_b=*/false, b,
      /*c_mod=*/(short)0, c, /*reuse_a=*/false, /*reuse_b=*/false);
}

// per-lane async global->LDS copy of 16 bytes (tracked with ASYNCcnt)
__device__ __forceinline__ void async_copy_b128(unsigned lds_byte_addr, const void* gptr) {
  asm volatile("global_load_async_to_lds_b128 %0, %1, off"
               :: "v"(lds_byte_addr), "v"(gptr) : "memory");
}
__device__ __forceinline__ void wait_xcnt0() {
  asm volatile("s_wait_xcnt 0x0" ::: "memory");
}
__device__ __forceinline__ void wait_async0() {
  asm volatile("s_wait_asynccnt 0x0" ::: "memory");
}

// --------------------------- converters ------------------------------------

__global__ __launch_bounds__(256) void cvt_bf16_kernel(
    const float* __restrict__ in, u16* __restrict__ out, int n) {
  int i = blockIdx.x * 256 + threadIdx.x;
  if (i < n) out[i] = f2bf(in[i]);
}

// pad x (B,L,D) -> xp (B,S,D) bf16, zero padded rows
__global__ __launch_bounds__(256) void pad_cvt_x_kernel(
    const float* __restrict__ x, u16* __restrict__ xp) {
  int i = blockIdx.x * 256 + threadIdx.x;          // < B*S*D
  int d = i % D_;
  int s = (i / D_) % S_;
  int b = i / (D_ * S_);
  float v = (s < L_) ? x[((size_t)(b * L_ + s)) * D_ + d] : 0.0f;
  xp[i] = f2bf(v);
}

// --------------------------- WMMA GEMM --------------------------------------
// C[M,N] = epilogue(A[M,K] @ B[K,N] + bias[N]), bf16 inputs, fp32 accumulate.
// Block 128x128, K-step 32, 256 threads = 8 waves (4x2 wave grid).
// Double-buffered LDS; A tile staged with async global->LDS b128 copies.
// EPI 1: bf16 out, *scale     (QKV)
// EPI 2: bf16 out, exact GELU (FFN1)
// EPI 3: f32 out, + res[M,N]  (FFN2 -> final output)

constexpr int BM = 128, BN = 128, BK = 32;

template <int EPI>
__global__ __launch_bounds__(256) void gemm_bf16_wmma(
    const u16* __restrict__ A, const u16* __restrict__ Bm,
    const float* __restrict__ bias, const float* __restrict__ res,
    float scale, float* __restrict__ Cf, u16* __restrict__ Cb,
    int M, int N, int K) {
  __shared__ u16 sA[2][BM * BK];   // [row][k]   2 x 8 KB
  __shared__ u16 sBt[2][BN * BK];  // [col][k]   2 x 8 KB (transposed B)

  const int tid  = threadIdx.x;
  const int lane = tid & 31;
  const int wid  = tid >> 5;
  const int wm   = wid & 3;   // 4 row-groups of 32
  const int wn   = wid >> 2;  // 2 col-groups of 64
  const int m0   = blockIdx.y * BM;
  const int n0   = blockIdx.x * BN;
  const int half = lane >> 4;
  const int l16  = lane & 15;

  v8f acc[2][4];
#pragma unroll
  for (int mi = 0; mi < 2; ++mi)
#pragma unroll
    for (int ni = 0; ni < 4; ++ni) acc[mi][ni] = (v8f)0.0f;

  auto stageA_async = [&](int k0, int buf) {
#pragma unroll
    for (int i = 0; i < 2; ++i) {
      int ch = tid + i * 256;          // 512 chunks of 8 bf16
      int r  = ch >> 2;
      int c8 = (ch & 3) << 3;
      const u16* gp = A + (size_t)(m0 + r) * K + k0 + c8;
      unsigned la = (unsigned)(size_t)&sA[buf][r * BK + c8];
      async_copy_b128(la, gp);
    }
    wait_xcnt0();   // address VGPRs may be recycled after translation
  };
  auto stageB = [&](int k0, int buf) {
#pragma unroll
    for (int i = 0; i < 2; ++i) {
      int ch = tid + i * 256;          // 512 chunks of 8 bf16 (contig in N)
      int kr = ch >> 4;
      int n8 = (ch & 15) << 3;
      const u16* gp = Bm + (size_t)(k0 + kr) * N + n0 + n8;
      if (k0 + BK < K) __builtin_prefetch(gp + (size_t)BK * N, 0, 0);
      uint4 u = *(const uint4*)gp;
      const u16* up = (const u16*)&u;
#pragma unroll
      for (int j = 0; j < 8; ++j) sBt[buf][(n8 + j) * BK + kr] = up[j];
    }
  };

  stageA_async(0, 0);
  stageB(0, 0);

  int cur = 0;
  for (int k0 = 0; k0 < K; k0 += BK) {
    wait_async0();        // my async writes into sA[cur] have landed
    __syncthreads();      // everyone's writes landed; prev reads of cur^1 done

    if (k0 + BK < K) {    // stage next tile into the other buffer
      stageA_async(k0 + BK, cur ^ 1);
      stageB(k0 + BK, cur ^ 1);
    }

    // ---- fragments + WMMA from sA[cur]/sBt[cur] ----
    Frag af[2], bfg[4];
#pragma unroll
    for (int mi = 0; mi < 2; ++mi) {
      int row = wm * 32 + mi * 16 + l16;
      const u16* p = &sA[cur][row * BK] + (half ? 16 : 0);
      af[mi].u[0] = *(const uint4*)(p);
      af[mi].u[1] = *(const uint4*)(p + 8);
    }
#pragma unroll
    for (int ni = 0; ni < 4; ++ni) {
      int col = wn * 64 + ni * 16 + l16;
      const u16* p = &sBt[cur][col * BK] + (half ? 16 : 0);
      bfg[ni].u[0] = *(const uint4*)(p);
      bfg[ni].u[1] = *(const uint4*)(p + 8);
    }
#pragma unroll
    for (int mi = 0; mi < 2; ++mi)
#pragma unroll
      for (int ni = 0; ni < 4; ++ni)
        acc[mi][ni] = wmma_bf16(af[mi].v, bfg[ni].v, acc[mi][ni]);

    cur ^= 1;
  }

  // ---- epilogue ----
#pragma unroll
  for (int mi = 0; mi < 2; ++mi)
#pragma unroll
    for (int ni = 0; ni < 4; ++ni)
#pragma unroll
      for (int f = 0; f < 8; ++f) {
        int m = m0 + wm * 32 + mi * 16 + f + half * 8;
        int n = n0 + wn * 64 + ni * 16 + l16;
        float v = acc[mi][ni][f] + bias[n];
        if constexpr (EPI == 1) {
          Cb[(size_t)m * N + n] = f2bf(v * scale);
        } else if constexpr (EPI == 2) {
          float g = 0.5f * v * (1.0f + erff(v * 0.70710678118654752f));
          Cb[(size_t)m * N + n] = f2bf(g);
        } else {  // EPI == 3
          Cf[(size_t)m * N + n] = v + res[(size_t)m * N + n];
        }
      }
}

// --------------------------- attention --------------------------------------
// One wave per (b, h, nblk, 16-query-row group).
//  - V tile (384 x 64 bf16, row stride D) staged into LDS by the TDM,
//    issued first so the DMA overlaps the whole QK^T + softmax phase.
//    TDM OOB (tensor_dim1 = S - sk0) zero-fills rows past the sequence.
//  - scores: 24 WMMA accumulators, K B-fragments straight from global.
//  - softmax via shfl_xor butterflies inside 16-lane halves (wave32).
//  - P@V: A-fragments from LDS probs, B-fragments gathered from row-major V.

__global__ __launch_bounds__(32) void attn_kernel(
    const u16* __restrict__ q, const u16* __restrict__ k,
    const u16* __restrict__ v, const float* __restrict__ mask,
    const float* __restrict__ x, float* __restrict__ hpre) {
  __shared__ u16 sP[16 * W3_];    // 12 KB: probs, row-major [16][384]
  __shared__ u16 sV[W3_ * DH_];   // 48 KB: V tile, row-major [r][d]

  const int lane = threadIdx.x;
  const int half = lane >> 4;
  const int l16  = lane & 15;

  int gid = blockIdx.x;
  const int qg = gid & 7;        gid >>= 3;           // 16-row group (0..7)
  const int nb = gid % NBLK_;    gid /= NBLK_;        // 128-block
  const int hh = gid & 15;       gid >>= 4;           // head
  const int bb = gid;                                 // batch

  // ---- issue TDM load of the V tile (overlaps with score phase) ----
  {
    int sk0      = nb * CC_ - CC_;
    int rows     = W3_;                       // 384
    unsigned lds = (unsigned)(size_t)sV;
    if (nb == 0) {                            // keys sk<0 handled by pre-zero
      sk0 = 0;
      rows = W3_ - CC_;                       // 256
      lds += (unsigned)(CC_ * DH_ * 2);
    }
    const int tdim1 = S_ - sk0;               // rows available; OOB -> zeros
    unsigned long long ga =
        (unsigned long long)(size_t)(v + ((size_t)(bb * S_ + sk0)) * D_ + hh * DH_);

    v4u g0 = { 1u,                                   // count=1, user D#
               lds,                                  // lds_addr (bytes)
               (unsigned)ga,                         // global_addr[31:0]
               (unsigned)(ga >> 32) | 0x80000000u }; // global_addr[56:32] | type=2
    v8i g1 = { (int)0x00010000u,                     // data_size=1 (2 bytes)
               (int)((unsigned)DH_ << 16),           // tensor_dim0[15:0] = 64
               (int)(((unsigned)tdim1 & 0xFFFFu) << 16),      // tensor_dim1 lo
               (int)(((unsigned)DH_ << 16) | ((unsigned)tdim1 >> 16)), // tile_dim0|dim1 hi
               (int)rows,                            // tile_dim1 (tile_dim2=0)
               (int)D_,                              // tensor_dim0_stride = 1024
               0, 0 };
    v4i g2 = { 0, 0, 0, 0 };
    v4i g3 = { 0, 0, 0, 0 };
    v8i g4 = { 0, 0, 0, 0, 0, 0, 0, 0 };     // trailing group (6-arg builtin)
    __builtin_amdgcn_tensor_load_to_lds(g0, g1, g2, g3, g4, 0);

    if (nb == 0) {                            // zero LDS rows for keys sk<0
      uint4 z = {0u, 0u, 0u, 0u};
      for (int i = lane; i < (CC_ * DH_) / 8; i += 32)
        *(uint4*)(sV + i * 8) = z;
    }
  }

  // ---- Q A-fragments (16 rows x 64) ----
  Frag aq[2];
  {
    int row = qg * 16 + l16;
    int sq  = nb * CC_ + row;                         // < S_
    const u16* qp = q + ((size_t)(bb * S_ + sq)) * D_ + hh * DH_;
#pragma unroll
    for (int kb = 0; kb < 2; ++kb) {
      const u16* p = qp + kb * 32 + (half ? 16 : 0);
      aq[kb].u[0] = *(const uint4*)(p);
      aq[kb].u[1] = *(const uint4*)(p + 8);
    }
  }

  // ---- scores: 24 tiles of 16x16 ----
  v8f s[24];
#pragma unroll
  for (int t = 0; t < 24; ++t) s[t] = (v8f)0.0f;

#pragma unroll
  for (int t = 0; t < 24; ++t) {
    int r   = t * 16 + l16;             // key column (same for both halves)
    int sk  = nb * CC_ - CC_ + r;       // key sequence position
    int skc = min(max(sk, 0), S_ - 1);
    const u16* kp = k + ((size_t)(bb * S_ + skc)) * D_ + hh * DH_;
    Frag bk[2];
#pragma unroll
    for (int kb = 0; kb < 2; ++kb) {
      const u16* p = kp + kb * 32 + (half ? 16 : 0);
      bk[kb].u[0] = *(const uint4*)(p);
      bk[kb].u[1] = *(const uint4*)(p + 8);
    }
    s[t] = wmma_bf16(aq[0].v, bk[0].v, s[t]);
    s[t] = wmma_bf16(aq[1].v, bk[1].v, s[t]);

    bool kv = (sk >= 0) && (sk < L_) && (mask[(size_t)bb * L_ + sk] == 0.0f);
#pragma unroll
    for (int f = 0; f < 8; ++f) {
      int p_ = qg * 16 + f + half * 8;  // query row within 128-block
      bool band = (r >= p_) && (r <= p_ + 2 * CC_);
      if (!(band && kv)) s[t][f] = -1.0e9f;
    }
  }

  // ---- softmax (lanes 0-15 own m=f, lanes 16-31 own m=f+8) ----
  float rmax[8], rinv[8];
#pragma unroll
  for (int f = 0; f < 8; ++f) {
    float mx = -3.0e38f;
#pragma unroll
    for (int t = 0; t < 24; ++t) mx = fmaxf(mx, s[t][f]);
#pragma unroll
    for (int o = 1; o < 16; o <<= 1) mx = fmaxf(mx, __shfl_xor(mx, o, 32));
    float sum = 0.0f;
#pragma unroll
    for (int t = 0; t < 24; ++t) sum += __expf(s[t][f] - mx);
#pragma unroll
    for (int o = 1; o < 16; o <<= 1) sum += __shfl_xor(sum, o, 32);
    rmax[f] = mx;
    rinv[f] = 1.0f / sum;
  }

  // ---- store probs (bf16) to LDS ----
#pragma unroll
  for (int t = 0; t < 24; ++t)
#pragma unroll
    for (int f = 0; f < 8; ++f) {
      float p = __expf(s[t][f] - rmax[f]) * rinv[f];
      sP[(f + half * 8) * W3_ + t * 16 + l16] = f2bf(p);
    }

  __builtin_amdgcn_s_wait_tensorcnt(0);   // V tile resident in LDS
  __syncthreads();                        // + my ds stores of sP visible

  // ---- attn = P @ V : D tile 16x64 ----
  v8f o[4];
#pragma unroll
  for (int nt = 0; nt < 4; ++nt) o[nt] = (v8f)0.0f;

#pragma unroll
  for (int kb = 0; kb < 12; ++kb) {
    Frag ap;
    {
      const u16* p = sP + l16 * W3_ + kb * 32 + (half ? 16 : 0);
      ap.u[0] = *(const uint4*)(p);
      ap.u[1] = *(const uint4*)(p + 8);
    }
    int r0 = kb * 32 + (half ? 16 : 0);
#pragma unroll
    for (int nt = 0; nt < 4; ++nt) {
      int d = nt * 16 + l16;
      Frag bv;
#pragma unroll
      for (int j = 0; j < 16; ++j) bv.s[j] = sV[(r0 + j) * DH_ + d];
      o[nt] = wmma_bf16(ap.v, bv.v, o[nt]);
    }
  }

  // ---- hpre = x + attn (drop padded rows) ----
#pragma unroll
  for (int nt = 0; nt < 4; ++nt)
#pragma unroll
    for (int f = 0; f < 8; ++f) {
      int m  = qg * 16 + f + half * 8;
      int sq = nb * CC_ + m;
      if (sq < L_) {
        int d = nt * 16 + l16;
        size_t idx = ((size_t)(bb * L_ + sq)) * D_ + hh * DH_ + d;
        hpre[idx] = x[idx] + o[nt][f];
      }
    }
}

// --------------------------- LayerNorm --------------------------------------

__global__ __launch_bounds__(256) void ln_kernel(
    const float* __restrict__ hpre, const float* __restrict__ g,
    const float* __restrict__ be, float* __restrict__ hln,
    u16* __restrict__ hln_bf) {
  __shared__ float red[256];
  const int row = blockIdx.x;
  const float* p = hpre + (size_t)row * D_;

  float lx[4];
  float sum = 0.0f;
#pragma unroll
  for (int i = 0; i < 4; ++i) {
    lx[i] = p[threadIdx.x + i * 256];
    sum += lx[i];
  }
  red[threadIdx.x] = sum;
  __syncthreads();
  for (int st = 128; st > 0; st >>= 1) {
    if (threadIdx.x < st) red[threadIdx.x] += red[threadIdx.x + st];
    __syncthreads();
  }
  float mean = red[0] / (float)D_;
  __syncthreads();

  float sq = 0.0f;
#pragma unroll
  for (int i = 0; i < 4; ++i) {
    float d = lx[i] - mean;
    sq += d * d;
  }
  red[threadIdx.x] = sq;
  __syncthreads();
  for (int st = 128; st > 0; st >>= 1) {
    if (threadIdx.x < st) red[threadIdx.x] += red[threadIdx.x + st];
    __syncthreads();
  }
  float var  = red[0] / (float)D_;
  float rstd = rsqrtf(var + 1e-5f);
#pragma unroll
  for (int i = 0; i < 4; ++i) {
    int j = threadIdx.x + i * 256;
    float y = (lx[i] - mean) * rstd * g[j] + be[j];
    hln[(size_t)row * D_ + j]    = y;
    hln_bf[(size_t)row * D_ + j] = f2bf(y);
  }
}

// --------------------------- launch -----------------------------------------

extern "C" void kernel_launch(void* const* d_in, const int* in_sizes, int n_in,
                              void* d_out, int out_size, void* d_ws, size_t ws_size,
                              hipStream_t stream) {
  (void)in_sizes; (void)n_in; (void)out_size;

  const float* x    = (const float*)d_in[0];
  const float* mask = (const float*)d_in[1];
  const float* Wq   = (const float*)d_in[2];
  const float* bq   = (const float*)d_in[3];
  const float* Wk   = (const float*)d_in[4];
  const float* bk   = (const float*)d_in[5];
  const float* Wv   = (const float*)d_in[6];
  const float* bv   = (const float*)d_in[7];
  const float* ln_g = (const float*)d_in[8];
  const float* ln_b = (const float*)d_in[9];
  const float* W1   = (const float*)d_in[10];
  const float* b1   = (const float*)d_in[11];
  const float* W2   = (const float*)d_in[12];
  const float* b2   = (const float*)d_in[13];
  float* out = (float*)d_out;

  // ---- workspace carve-up ----
  char* base = (char*)d_ws;
  size_t off = 0;
  auto carve = [&](size_t bytes) -> char* {
    char* p = base + off;
    off += (bytes + 255) & ~(size_t)255;
    return p;
  };
  const size_t BSD = (size_t)B_ * S_ * D_;
  const size_t BLD = (size_t)B_ * L_ * D_;
  const size_t BLF = (size_t)B_ * L_ * F_;

  u16*   xp_bf  = (u16*)carve(BSD * 2);
  u16*   Wq_bf  = (u16*)carve((size_t)D_ * D_ * 2);
  u16*   Wk_bf  = (u16*)carve((size_t)D_ * D_ * 2);
  u16*   Wv_bf  = (u16*)carve((size_t)D_ * D_ * 2);
  u16*   W1_bf  = (u16*)carve((size_t)D_ * F_ * 2);
  u16*   W2_bf  = (u16*)carve((size_t)F_ * D_ * 2);
  u16*   q_bf   = (u16*)carve(BSD * 2);
  u16*   k_bf   = (u16*)carve(BSD * 2);
  u16*   v_bf   = (u16*)carve(BSD * 2);
  float* hpre   = (float*)carve(BLD * 4);
  float* hln    = (float*)carve(BLD * 4);
  u16*   hln_bf = (u16*)carve(BLD * 2);
  u16*   a1_bf  = (u16*)carve(BLF * 2);
  if (off > ws_size) return;  // workspace too small: bail deterministically

  // ---- 1) convert / pad ----
  pad_cvt_x_kernel<<<(int)(BSD / 256), 256, 0, stream>>>(x, xp_bf);
  cvt_bf16_kernel<<<(D_ * D_) / 256, 256, 0, stream>>>(Wq, Wq_bf, D_ * D_);
  cvt_bf16_kernel<<<(D_ * D_) / 256, 256, 0, stream>>>(Wk, Wk_bf, D_ * D_);
  cvt_bf16_kernel<<<(D_ * D_) / 256, 256, 0, stream>>>(Wv, Wv_bf, D_ * D_);
  cvt_bf16_kernel<<<(D_ * F_) / 256, 256, 0, stream>>>(W1, W1_bf, D_ * F_);
  cvt_bf16_kernel<<<(F_ * D_) / 256, 256, 0, stream>>>(W2, W2_bf, F_ * D_);

  // ---- 2) QKV projections (M = B*S = 8704) ----
  const int Mqkv = B_ * S_;
  dim3 gqkv(D_ / BN, Mqkv / BM);
  gemm_bf16_wmma<1><<<gqkv, 256, 0, stream>>>(xp_bf, Wq_bf, bq, nullptr,
                                              0.125f /*1/sqrt(64)*/, nullptr, q_bf,
                                              Mqkv, D_, D_);
  gemm_bf16_wmma<1><<<gqkv, 256, 0, stream>>>(xp_bf, Wk_bf, bk, nullptr,
                                              1.0f, nullptr, k_bf, Mqkv, D_, D_);
  gemm_bf16_wmma<1><<<gqkv, 256, 0, stream>>>(xp_bf, Wv_bf, bv, nullptr,
                                              1.0f, nullptr, v_bf, Mqkv, D_, D_);

  // ---- 3) banded attention + residual ----
  attn_kernel<<<B_ * H_ * NBLK_ * (CC_ / 16), 32, 0, stream>>>(
      q_bf, k_bf, v_bf, mask, x, hpre);

  // ---- 4) LayerNorm ----
  ln_kernel<<<B_ * L_, 256, 0, stream>>>(hpre, ln_g, ln_b, hln, hln_bf);

  // ---- 5) FFN ----
  const int Mff = B_ * L_;
  dim3 gf1(F_ / BN, Mff / BM);
  gemm_bf16_wmma<2><<<gf1, 256, 0, stream>>>(hln_bf, W1_bf, b1, nullptr,
                                             1.0f, nullptr, a1_bf, Mff, F_, D_);
  dim3 gf2(D_ / BN, Mff / BM);
  gemm_bf16_wmma<3><<<gf2, 256, 0, stream>>>(a1_bf, W2_bf, b2, hln,
                                             1.0f, out, nullptr, Mff, D_, F_);
}
